// MultiQueryAttention_61813169324769
// MI455X (gfx1250) — matrix operand order repaired and checked by reading, not compiled, not run on hardware
//
#include <hip/hip_runtime.h>

typedef __attribute__((ext_vector_type(16))) __bf16 v16bf;
typedef __attribute__((ext_vector_type(8)))  __bf16 v8bf;
typedef __attribute__((ext_vector_type(8)))  float  v8f;

#define BATCH 2
#define SEQ   2048
#define NH    16
#define KD    64
#define VD    64

// ---------------------------------------------------------------------------
// Kernel A: K = x @ key_proj (bf16, [B][M][64] row-major)
//           V = x @ value_proj stored transposed (bf16, [B][64][M])
// ---------------------------------------------------------------------------
__global__ void proj_kv(const float* __restrict__ x,
                        const float* __restrict__ kp,
                        const float* __restrict__ vp,
                        __bf16* __restrict__ Kbf,
                        __bf16* __restrict__ Vt) {
  int idx = blockIdx.x * blockDim.x + threadIdx.x;   // B*SEQ*64 threads
  int i = idx & 63;
  int m = (idx >> 6) & (SEQ - 1);
  int b = idx >> 17;
  const float* xr = x + ((size_t)b * SEQ + m) * KD;
  float ks = 0.f, vs = 0.f;
#pragma unroll 8
  for (int d = 0; d < KD; ++d) {
    float xv = xr[d];
    ks += xv * kp[d * KD + i];
    vs += xv * vp[d * VD + i];
  }
  Kbf[((size_t)b * SEQ + m) * KD + i] = (__bf16)ks;
  Vt[((size_t)b * VD + i) * SEQ + m]  = (__bf16)vs;
}

// ---------------------------------------------------------------------------
// Kernel B: Q[b,h,n,k] = sum_d x[b,n,d]*query_proj[h,k,d], pre-scaled by 1/8
// stored bf16 [B][H][SEQ][64]
// ---------------------------------------------------------------------------
__global__ void proj_q(const float* __restrict__ x,
                       const float* __restrict__ qp,
                       __bf16* __restrict__ Qbf) {
  int idx = blockIdx.x * blockDim.x + threadIdx.x;   // B*H*SEQ*64 = 2^22
  int k = idx & 63;
  int n = (idx >> 6) & (SEQ - 1);
  int h = (idx >> 17) & (NH - 1);
  int b = idx >> 21;
  const float* xr = x + ((size_t)b * SEQ + n) * KD;
  const float* qr = qp + ((size_t)h * KD + k) * KD;
  float s = 0.f;
#pragma unroll 8
  for (int d = 0; d < KD; ++d) s += xr[d] * qr[d];
  Qbf[(((size_t)(b * NH + h)) * SEQ + n) * KD + k] = (__bf16)(s * 0.125f);
}

// ---------------------------------------------------------------------------
// Kernel C: flash attention. 4 waves/block, each wave owns 16 query rows.
// Streams 64 keys per iteration: 8 WMMAs for S (4 n-tiles x 2 k-chunks),
// one online-softmax update (butterfly cost amortized over 64 keys), LDS
// bounce for P (C-layout -> A-layout), 8 WMMAs for O += P*V.
// ---------------------------------------------------------------------------
__global__ void __launch_bounds__(128) mqa_attn(const __bf16* __restrict__ Qbf,
                                                const __bf16* __restrict__ Kbf,
                                                const __bf16* __restrict__ Vt,
                                                float* __restrict__ O) {
  const int lane = threadIdx.x & 31;
  const int wave = threadIdx.x >> 5;
  const int qt = blockIdx.x & 31;          // N/64 tiles
  const int h  = (blockIdx.x >> 5) & (NH - 1);
  const int b  = blockIdx.x >> 9;
  const int q0 = qt * 64 + wave * 16;

  const int hl  = lane >> 4;               // half of wave (0/1)
  const int l16 = lane & 15;
  const int abase = hl * 16;               // A-layout K base for this lane

  // ---- A-fragments of Q (fixed for the whole key loop) ----
  const __bf16* Qrow = Qbf + (((size_t)(b * NH + h)) * SEQ + q0 + l16) * KD;
  union F { v16bf v; v8bf p[2]; };
  F a0, a1, pa0, pa1;
  a0.p[0] = *(const v8bf*)(Qrow + abase);            // K = base..base+7
  a0.p[1] = *(const v8bf*)(Qrow + abase + 8);        // K = base+8..base+15
  a1.p[0] = *(const v8bf*)(Qrow + 32 + abase);       // K = 32+...
  a1.p[1] = *(const v8bf*)(Qrow + 32 + abase + 8);

  v8f o0 = {}, o1 = {}, o2 = {}, o3 = {};
  float mrow[8], lrow[8];
#pragma unroll
  for (int i = 0; i < 8; ++i) { mrow[i] = -1e30f; lrow[i] = 0.f; }

  __shared__ __align__(32) __bf16 Pl[4][16][64];
  __bf16 (*P)[64] = Pl[wave];

  const __bf16* Kb = Kbf + (size_t)b * SEQ * KD;
  const __bf16* Vb = Vt  + (size_t)b * VD * SEQ;

  for (int n0 = 0; n0 < SEQ; n0 += 64) {
    // ---- S = Q * K^T for four 16-key tiles (B-frag: 16 contiguous bf16/lane)
    v8f s0 = {}, s1 = {}, s2 = {}, s3 = {};
    {
      const __bf16* kr = Kb + (size_t)(n0 + l16) * KD + hl * 16;
      v16bf bk;
      bk = *(const v16bf*)(kr);
      s0 = __builtin_amdgcn_wmma_f32_16x16x32_bf16(false, a0.v, false, bk, (short)0, s0, false, false);
      bk = *(const v16bf*)(kr + 32);
      s0 = __builtin_amdgcn_wmma_f32_16x16x32_bf16(false, a1.v, false, bk, (short)0, s0, false, false);
      bk = *(const v16bf*)(kr + 16 * KD);
      s1 = __builtin_amdgcn_wmma_f32_16x16x32_bf16(false, a0.v, false, bk, (short)0, s1, false, false);
      bk = *(const v16bf*)(kr + 16 * KD + 32);
      s1 = __builtin_amdgcn_wmma_f32_16x16x32_bf16(false, a1.v, false, bk, (short)0, s1, false, false);
      bk = *(const v16bf*)(kr + 32 * KD);
      s2 = __builtin_amdgcn_wmma_f32_16x16x32_bf16(false, a0.v, false, bk, (short)0, s2, false, false);
      bk = *(const v16bf*)(kr + 32 * KD + 32);
      s2 = __builtin_amdgcn_wmma_f32_16x16x32_bf16(false, a1.v, false, bk, (short)0, s2, false, false);
      bk = *(const v16bf*)(kr + 48 * KD);
      s3 = __builtin_amdgcn_wmma_f32_16x16x32_bf16(false, a0.v, false, bk, (short)0, s3, false, false);
      bk = *(const v16bf*)(kr + 48 * KD + 32);
      s3 = __builtin_amdgcn_wmma_f32_16x16x32_bf16(false, a1.v, false, bk, (short)0, s3, false, false);
    }

    // ---- one online-softmax update for all 64 key columns
    float p0[8], p1[8], p2[8], p3[8];
#pragma unroll
    for (int i = 0; i < 8; ++i) {
      float mx = fmaxf(fmaxf(s0[i], s1[i]), fmaxf(s2[i], s3[i]));
      mx = fmaxf(mx, __shfl_xor(mx, 1, 32));
      mx = fmaxf(mx, __shfl_xor(mx, 2, 32));
      mx = fmaxf(mx, __shfl_xor(mx, 4, 32));
      mx = fmaxf(mx, __shfl_xor(mx, 8, 32));
      float mnew = fmaxf(mrow[i], mx);
      float cf = __expf(mrow[i] - mnew);
      mrow[i] = mnew;
      p0[i] = __expf(s0[i] - mnew);
      p1[i] = __expf(s1[i] - mnew);
      p2[i] = __expf(s2[i] - mnew);
      p3[i] = __expf(s3[i] - mnew);
      float rs = (p0[i] + p1[i]) + (p2[i] + p3[i]);
      rs += __shfl_xor(rs, 1, 32);
      rs += __shfl_xor(rs, 2, 32);
      rs += __shfl_xor(rs, 4, 32);
      rs += __shfl_xor(rs, 8, 32);
      lrow[i] = lrow[i] * cf + rs;
      o0[i] *= cf; o1[i] *= cf; o2[i] *= cf; o3[i] *= cf;
    }

    // ---- bounce P through LDS: C-layout (m=i+8*hl, n=l16+16t) -> A-layout
#pragma unroll
    for (int i = 0; i < 8; ++i) {
      int r = i + 8 * hl;
      P[r][l16]      = (__bf16)p0[i];
      P[r][l16 + 16] = (__bf16)p1[i];
      P[r][l16 + 32] = (__bf16)p2[i];
      P[r][l16 + 48] = (__bf16)p3[i];
    }
    asm volatile("s_wait_dscnt 0x0" ::: "memory");
    pa0.p[0] = *(const v8bf*)&P[l16][abase];           // keys 0..31 of block
    pa0.p[1] = *(const v8bf*)&P[l16][abase + 8];
    pa1.p[0] = *(const v8bf*)&P[l16][32 + abase];      // keys 32..63 of block
    pa1.p[1] = *(const v8bf*)&P[l16][32 + abase + 8];

    // ---- O += P * V  (V^T rows give contiguous B-frags: keys n0+16*hl+e)
    const __bf16* vcol = Vb + (size_t)l16 * SEQ + n0 + hl * 16;
    v16bf bv;
    bv = *(const v16bf*)(vcol);
    o0 = __builtin_amdgcn_wmma_f32_16x16x32_bf16(false, pa0.v, false, bv, (short)0, o0, false, false);
    bv = *(const v16bf*)(vcol + 32);
    o0 = __builtin_amdgcn_wmma_f32_16x16x32_bf16(false, pa1.v, false, bv, (short)0, o0, false, false);
    bv = *(const v16bf*)(vcol + 16 * SEQ);
    o1 = __builtin_amdgcn_wmma_f32_16x16x32_bf16(false, pa0.v, false, bv, (short)0, o1, false, false);
    bv = *(const v16bf*)(vcol + 16 * SEQ + 32);
    o1 = __builtin_amdgcn_wmma_f32_16x16x32_bf16(false, pa1.v, false, bv, (short)0, o1, false, false);
    bv = *(const v16bf*)(vcol + 32 * SEQ);
    o2 = __builtin_amdgcn_wmma_f32_16x16x32_bf16(false, pa0.v, false, bv, (short)0, o2, false, false);
    bv = *(const v16bf*)(vcol + 32 * SEQ + 32);
    o2 = __builtin_amdgcn_wmma_f32_16x16x32_bf16(false, pa1.v, false, bv, (short)0, o2, false, false);
    bv = *(const v16bf*)(vcol + 48 * SEQ);
    o3 = __builtin_amdgcn_wmma_f32_16x16x32_bf16(false, pa0.v, false, bv, (short)0, o3, false, false);
    bv = *(const v16bf*)(vcol + 48 * SEQ + 32);
    o3 = __builtin_amdgcn_wmma_f32_16x16x32_bf16(false, pa1.v, false, bv, (short)0, o3, false, false);
  }

  // ---- epilogue: normalize and store O[b][n][h][v] fp32
#pragma unroll
  for (int i = 0; i < 8; ++i) {
    float inv = 1.0f / lrow[i];
    int n = q0 + i + 8 * hl;
    float* orow = O + (((size_t)(b * SEQ + n)) * NH + h) * VD + l16;
    orow[0]  = o0[i] * inv;
    orow[16] = o1[i] * inv;
    orow[32] = o2[i] * inv;
    orow[48] = o3[i] * inv;
  }
}

// ---------------------------------------------------------------------------
// Kernel D: out[b,n,d] = sum_{h,v} O[b,n,h,v] * output_proj[d,h,v]
// ---------------------------------------------------------------------------
__global__ void out_proj(const float* __restrict__ Oc,
                         const float* __restrict__ op,
                         float* __restrict__ out) {
  int idx = blockIdx.x * blockDim.x + threadIdx.x;   // B*SEQ*64 threads
  int d = idx & 63;
  int n = (idx >> 6) & (SEQ - 1);
  int b = idx >> 17;
  const float* orow = Oc + ((size_t)(b * SEQ + n)) * (NH * VD);
  const float* prow = op + (size_t)d * (NH * VD);
  float s = 0.f;
#pragma unroll 8
  for (int j = 0; j < NH * VD; ++j) s += orow[j] * prow[j];
  out[idx] = s;
}

// ---------------------------------------------------------------------------
extern "C" void kernel_launch(void* const* d_in, const int* in_sizes, int n_in,
                              void* d_out, int out_size, void* d_ws, size_t ws_size,
                              hipStream_t stream) {
  (void)in_sizes; (void)n_in; (void)out_size; (void)ws_size;
  const float* x  = (const float*)d_in[0];
  const float* qp = (const float*)d_in[1];
  const float* kp = (const float*)d_in[2];
  const float* vp = (const float*)d_in[3];
  const float* op = (const float*)d_in[4];
  float* out = (float*)d_out;

  // workspace layout (bf16 elements, then f32 O)
  __bf16* Qbf = (__bf16*)d_ws;                                   // B*H*SEQ*64
  __bf16* Kbf = Qbf + (size_t)BATCH * NH * SEQ * KD;             // B*SEQ*64
  __bf16* Vt  = Kbf + (size_t)BATCH * SEQ * KD;                  // B*64*SEQ
  float*  O   = (float*)(Vt + (size_t)BATCH * VD * SEQ);         // B*SEQ*H*64

  proj_kv<<<(BATCH * SEQ * KD) / 256, 256, 0, stream>>>(x, kp, vp, Kbf, Vt);
  proj_q<<<(BATCH * NH * SEQ * KD) / 256, 256, 0, stream>>>(x, qp, Qbf);
  mqa_attn<<<BATCH * NH * (SEQ / 64), 128, 0, stream>>>(Qbf, Kbf, Vt, O);
  out_proj<<<(BATCH * SEQ * KD) / 256, 256, 0, stream>>>(O, op, out);
}
